// sphericalConv_13443247636948
// MI455X (gfx1250) — compile-verified, matrix operand order checked by reading, not run on hardware
//
#include <hip/hip_runtime.h>

#define NSIDE 128
#define NPIX (12 * NSIDE * NSIDE)   // 196608
#define BATCH 8
#define CIN 16
#define COUT 32
#define PIX_PER_BLOCK 64
#define THREADS 128

typedef __attribute__((ext_vector_type(16))) __bf16 v16bf;
typedef __attribute__((ext_vector_type(8)))  float  v8f;

union BF16x16 {
    v16bf v;
    uint4 q[2];
};

// ---------------------------------------------------------------------------
// Kernel 1: transpose + f32->bf16 convert:  x(B,CIN,NPIX) -> xT(B,NPIX+1,CIN)
// Coalesced reads (256B per channel row), coalesced 16B-chunk writes.
// ---------------------------------------------------------------------------
__global__ void __launch_bounds__(THREADS)
prep_transpose(const float* __restrict__ x, __bf16* __restrict__ xT) {
    __shared__ __align__(16) __bf16 tile[PIX_PER_BLOCK][CIN];   // [64][16] = 2KB
    const int nblk = NPIX / PIX_PER_BLOCK;
    const int b     = blockIdx.x / nblk;
    const int pbase = (blockIdx.x % nblk) * PIX_PER_BLOCK;
    const int t     = threadIdx.x;

#pragma unroll
    for (int i = 0; i < (CIN * PIX_PER_BLOCK) / THREADS; ++i) {  // 8 iters
        const int e = i * THREADS + t;
        const int c = e >> 6;     // /64
        const int p = e & 63;
        const float v = x[((size_t)(b * CIN + c)) * NPIX + pbase + p];
        tile[p][c] = (__bf16)v;
    }
    __syncthreads();

    // 64 pixel rows * 32B each = 128 x 16B chunks, one per thread.
    const int px = t >> 1, half = t & 1;
    const uint4* src = (const uint4*)&tile[px][half * 8];
    uint4* dst = (uint4*)(xT + ((size_t)b * (NPIX + 1) + pbase + px) * CIN + half * 8);
    *dst = *src;
}

// ---------------------------------------------------------------------------
// Kernel 2: zero the pad row xT[b][NPIX][*]  (index NPIX => zeros)
// ---------------------------------------------------------------------------
__global__ void pad_zero(__bf16* __restrict__ xT) {
    const int t = threadIdx.x;                 // 16 threads
    const int b = t >> 1, half = t & 1;
    uint4* dst = (uint4*)(xT + ((size_t)b * (NPIX + 1) + NPIX) * CIN + half * 8);
    *dst = make_uint4(0u, 0u, 0u, 0u);
}

// ---------------------------------------------------------------------------
// Kernel 3: pack weights into per-lane WMMA B-fragment layout.
// Bpack layout: [step s(5)][ntile(2)][lane(32)][elem(16)] bf16.
// Lane l (g=l>>4, n=l&15) holds column N=n; elem e<8 -> K=8g+e, else K=16+8g+(e-8).
// Global K index kidx = neighbor*16 + channel; kidx >= 144 -> 0 (K padding).
// ---------------------------------------------------------------------------
__global__ void pack_weights(const float* __restrict__ w, __bf16* __restrict__ Bpack) {
    const int t = blockIdx.x * blockDim.x + threadIdx.x;
    if (t >= 5 * 2 * 32) return;
    const int lane = t & 31;
    const int nt   = (t >> 5) & 1;
    const int s    = t / 64;
    const int g = lane >> 4, n = lane & 15;
    const int o = nt * 16 + n;
#pragma unroll
    for (int e = 0; e < 16; ++e) {
        const int kk    = (e < 8) ? (8 * g + e) : (16 + 8 * g + (e - 8));
        const int kglob = s * 32 + kk;           // 0..159
        const int k = kglob >> 4;                // neighbor index
        const int c = kglob & 15;                // channel
        const float val = (k < 9) ? w[(o * CIN + c) * 9 + k] : 0.0f;
        Bpack[(size_t)t * 16 + e] = (__bf16)val;
    }
}

// ---------------------------------------------------------------------------
// Kernel 4: main gathered GEMM.
// Block = 4 wave32s = 64 pixels of one batch. Each wave: 16-pixel M-tile,
// 5 K-steps of v_wmma_f32_16x16x32_bf16 x 2 N-tiles, LDS-staged coalesced out.
// Gather offsets are 32-bit BYTE offsets so loads use SGPR-base + VGPR-offset
// addressing with a single v_lshl_or_b32 per neighbor (no 64-bit VALU chain).
// ---------------------------------------------------------------------------
__global__ void __launch_bounds__(THREADS)
conv_kernel(const __bf16* __restrict__ xT, const __bf16* __restrict__ Bpack,
            const int* __restrict__ neigh, const float* __restrict__ bias,
            float* __restrict__ out) {
    __shared__ __align__(16) float so[COUT][PIX_PER_BLOCK];   // 8 KB
    const int nblk   = NPIX / PIX_PER_BLOCK;
    const int b      = blockIdx.x / nblk;
    const int pblock = (blockIdx.x % nblk) * PIX_PER_BLOCK;
    const int t    = threadIdx.x;
    const int wave = t >> 5;
    const int lane = t & 31;
    const unsigned g = (unsigned)(lane >> 4);   // half-wave group
    const int m = lane & 15;                    // row within M-tile / N within D
    const unsigned p = (unsigned)(pblock + wave * 16 + m);

    int idx[9];
#pragma unroll
    for (int k = 0; k < 9; ++k) idx[k] = neigh[p * 9u + (unsigned)k];

    // Per-batch base: uniform across the block -> SGPR pair.
    const char* xb = (const char*)(xT + (size_t)b * (NPIX + 1) * CIN);
    const unsigned gbyte = g * 16u;             // which 16B half of the 32B row

    v8f acc0 = {};
    v8f acc1 = {};
#pragma unroll
    for (int s = 0; s < 5; ++s) {
        const unsigned n0 = (unsigned)idx[2 * s];
        const unsigned n1 = (s < 4) ? (unsigned)idx[2 * s + 1] : (unsigned)NPIX;

        // 32-bit byte offsets (max ~6.3MB): n<<5 | g<<4 -> one v_lshl_or_b32,
        // load becomes global_load_b128 vdst, voff, s[base:base+1].
        const unsigned byte0 = (n0 << 5) | gbyte;
        const unsigned byte1 = (n1 << 5) | gbyte;

        BF16x16 a;
        a.q[0] = *(const uint4*)(xb + byte0);   // K 0-7  / 8-15  (neighbor 2s)
        a.q[1] = *(const uint4*)(xb + byte1);   // K 16-23/ 24-31 (neighbor 2s+1)

        BF16x16 bf0, bf1;
        const unsigned boff = ((unsigned)(s * 2) * 32u + (unsigned)lane) * 16u;
        const uint4* bp0 = (const uint4*)(Bpack + boff);
        const uint4* bp1 = (const uint4*)(Bpack + boff + 32u * 16u);
        bf0.q[0] = bp0[0]; bf0.q[1] = bp0[1];
        bf1.q[0] = bp1[0]; bf1.q[1] = bp1[1];

        acc0 = __builtin_amdgcn_wmma_f32_16x16x32_bf16(
            false, a.v, false, bf0.v, (short)0, acc0, false, false);
        acc1 = __builtin_amdgcn_wmma_f32_16x16x32_bf16(
            false, a.v, false, bf1.v, (short)0, acc1, false, false);
    }

    const float bias0 = bias[m];
    const float bias1 = bias[m + 16];

    // D layout: VGPR r, lanes 0-15 -> M=r, N=lane; lanes 16-31 -> M=r+8, N=lane-16.
#pragma unroll
    for (int r = 0; r < 8; ++r) {
        const int mm = r + 8 * (int)g;
        so[m][wave * 16 + mm]      = acc0[r] + bias0;   // o = m
        so[m + 16][wave * 16 + mm] = acc1[r] + bias1;   // o = m + 16
    }
    __syncthreads();

    // Coalesced stores: 32 channels x 64 px = 512 float4 chunks.
    float* ob = out + ((size_t)b * COUT) * NPIX + pblock;
#pragma unroll
    for (int i = 0; i < 4; ++i) {
        const int e = i * THREADS + t;   // 0..511
        const int o = e >> 4;
        const int q = e & 15;
        const float4 v = *(const float4*)&so[o][q * 4];
        *(float4*)(ob + (size_t)o * NPIX + q * 4) = v;
    }
}

// ---------------------------------------------------------------------------
extern "C" void kernel_launch(void* const* d_in, const int* in_sizes, int n_in,
                              void* d_out, int out_size, void* d_ws, size_t ws_size,
                              hipStream_t stream) {
    const float* x    = (const float*)d_in[0];   // (B, CIN, NPIX) f32
    const float* w    = (const float*)d_in[1];   // (COUT, CIN, 9) f32
    const float* bias = (const float*)d_in[2];   // (COUT,) f32
    const int*   nb   = (const int*)d_in[3];     // (9*NPIX,) i32
    float* out = (float*)d_out;                  // (B, COUT, NPIX) f32

    __bf16* xT = (__bf16*)d_ws;
    const size_t xT_bytes = (size_t)BATCH * (NPIX + 1) * CIN * sizeof(__bf16);
    const size_t off = (xT_bytes + 255) & ~(size_t)255;
    __bf16* Bpack = (__bf16*)((char*)d_ws + off);

    const int nblk = NPIX / PIX_PER_BLOCK;   // 3072
    prep_transpose<<<BATCH * nblk, THREADS, 0, stream>>>(x, xT);
    pad_zero<<<1, 16, 0, stream>>>(xT);
    pack_weights<<<1, 320, 0, stream>>>(w, Bpack);
    conv_kernel<<<BATCH * nblk, THREADS, 0, stream>>>(xT, Bpack, nb, bias, out);
}
